// PointNetSetAbstraction_15238543966693
// MI455X (gfx1250) — matrix operand order repaired and checked by reading, not compiled, not run on hardware
//
#include <hip/hip_runtime.h>
#include <cstddef>

typedef __attribute__((ext_vector_type(16))) _Float16 v16h;
typedef __attribute__((ext_vector_type(8)))  _Float16 v8h;
typedef __attribute__((ext_vector_type(8)))  float    v8f;

#define NPTS   8192
#define BATCH  8
#define NPOINT 512
#define NSAMP  32
#define TOTCOL (NPOINT * NSAMP)   // 16384 columns per batch
#define RAD2   0.04f
#define BNEPS  1e-5f

// ---------------------------------------------------------------------------
// FPS: one 1024-thread block per batch. Points + running min-dist in registers.
// ---------------------------------------------------------------------------
__device__ __forceinline__ int block_argmax_1024(float v, int i, float* swv,
                                                 int* swi, int* s_res) {
  const int tid = threadIdx.x, lane = tid & 31, wv = tid >> 5;
#pragma unroll
  for (int d = 16; d > 0; d >>= 1) {
    float ov = __shfl_xor(v, d, 32);
    int   oi = __shfl_xor(i, d, 32);
    if (ov > v || (ov == v && oi < i)) { v = ov; i = oi; }
  }
  if (lane == 0) { swv[wv] = v; swi[wv] = i; }
  __syncthreads();
  if (wv == 0) {
    float v2 = swv[lane]; int i2 = swi[lane];
#pragma unroll
    for (int d = 16; d > 0; d >>= 1) {
      float ov = __shfl_xor(v2, d, 32);
      int   oi = __shfl_xor(i2, d, 32);
      if (ov > v2 || (ov == v2 && oi < i2)) { v2 = ov; i2 = oi; }
    }
    if (lane == 0) *s_res = i2;
  }
  __syncthreads();
  return *s_res;
}

__global__ void __launch_bounds__(1024)
fps_kernel(const float* __restrict__ xyz, int* __restrict__ fps_idx,
           float* __restrict__ newxyz /* (B,3,S) == d_out head */) {
  const int b = blockIdx.x;
  const int tid = threadIdx.x;
  const float* X = xyz + (size_t)b * 3 * NPTS;

  float px[8], py[8], pz[8], dr[8];
#pragma unroll
  for (int j = 0; j < 8; j++) {
    int n = tid + j * 1024;
    px[j] = X[n];
    py[j] = X[NPTS + n];
    pz[j] = X[2 * NPTS + n];
    dr[j] = 1e10f;
  }

  __shared__ float swv[32];
  __shared__ int   swi[32];
  __shared__ float s_c[3];
  __shared__ int   s_far;

  float bv = -3.4e38f; int bi = 0;
#pragma unroll
  for (int j = 0; j < 8; j++) {
    int n = tid + j * 1024;
    if (px[j] > bv) { bv = px[j]; bi = n; }
  }
  int far = block_argmax_1024(bv, bi, swv, swi, &s_far);

  for (int it = 0; it < NPOINT; it++) {
    if (tid == 0) {
      fps_idx[b * NPOINT + it] = far;
      float cx = X[far], cy = X[NPTS + far], cz = X[2 * NPTS + far];
      s_c[0] = cx; s_c[1] = cy; s_c[2] = cz;
      newxyz[(b * 3 + 0) * NPOINT + it] = cx;
      newxyz[(b * 3 + 1) * NPOINT + it] = cy;
      newxyz[(b * 3 + 2) * NPOINT + it] = cz;
    }
    __syncthreads();
    float cx = s_c[0], cy = s_c[1], cz = s_c[2];
    bv = -3.4e38f; bi = 0;
#pragma unroll
    for (int j = 0; j < 8; j++) {
      float dx = px[j] - cx, dy = py[j] - cy, dz = pz[j] - cz;
      float d = dx * dx + dy * dy + dz * dz;
      float dn = fminf(dr[j], d);
      dr[j] = dn;
      int n = tid + j * 1024;
      if (dn > bv) { bv = dn; bi = n; }
    }
    far = block_argmax_1024(bv, bi, swv, swi, &s_far);
  }
}

// ---------------------------------------------------------------------------
// Ball query: one wave per (b,s); first 32 in-radius indices in index order
// via wave32 ballot + prefix popcount.
// ---------------------------------------------------------------------------
__global__ void __launch_bounds__(256)
ballquery_kernel(const float* __restrict__ xyz,
                 const float* __restrict__ newxyz, int* __restrict__ gidx) {
  const int wvb  = threadIdx.x >> 5;
  const int lane = threadIdx.x & 31;
  const int wave = blockIdx.x * 8 + wvb;
  const int s = wave & (NPOINT - 1);
  const int b = wave >> 9;

  __shared__ int sbuf[8][NSAMP];

  const float cx = newxyz[(b * 3 + 0) * NPOINT + s];
  const float cy = newxyz[(b * 3 + 1) * NPOINT + s];
  const float cz = newxyz[(b * 3 + 2) * NPOINT + s];
  const float* X = xyz + (size_t)b * 3 * NPTS;

  int count = 0;
  for (int base = 0; base < NPTS && count < NSAMP; base += 32) {
    int n = base + lane;
    float dx = X[n] - cx, dy = X[NPTS + n] - cy, dz = X[2 * NPTS + n] - cz;
    bool pred = (dx * dx + dy * dy + dz * dz) <= RAD2;
    unsigned mask = (unsigned)__ballot(pred);
    int before = __popc(mask & ((1u << lane) - 1u));
    int pos = count + before;
    if (pred && pos < NSAMP) sbuf[wvb][pos] = n;
    count += __popc(mask);
  }
  if (count > NSAMP) count = NSAMP;
  __syncthreads();
  int first = (count > 0) ? sbuf[wvb][0] : 0;
  int v = (lane < count) ? sbuf[wvb][lane] : first;
  gidx[((size_t)(b * NPOINT + s)) * NSAMP + lane] = v;
}

// ---------------------------------------------------------------------------
// Pack W (Cout,Cin) f32 -> (Cout,Kpad) f16, zero-padded K.
// ---------------------------------------------------------------------------
__global__ void packW_kernel(const float* __restrict__ W,
                             _Float16* __restrict__ Wh, int Cout, int Cin,
                             int Kpad) {
  int i = blockIdx.x * 256 + threadIdx.x;
  if (i < Cout * Kpad) {
    int o = i / Kpad, c = i % Kpad;
    Wh[i] = (c < Cin) ? (_Float16)W[o * Cin + c] : (_Float16)0.0f;
  }
}

// ---------------------------------------------------------------------------
// Gather: build padded 96-channel f16 feature column per (b,s,k) sample.
// Layout: feat[(b*TOTCOL + col) * 96 + c]  (K-contiguous for b128 B-loads).
// ---------------------------------------------------------------------------
__global__ void __launch_bounds__(256)
gather_feat(const float* __restrict__ xyz, const float* __restrict__ pts,
            const int* __restrict__ gidx, const float* __restrict__ newxyz,
            _Float16* __restrict__ feat) {
  int gid = blockIdx.x * 256 + threadIdx.x;   // = b*16384 + s*32 + kk
  int s = (gid >> 5) & (NPOINT - 1);
  int b = gid >> 14;
  int id = gidx[gid];
  const float* Xz = xyz + (size_t)b * 3 * NPTS;
  const float* P  = pts + (size_t)b * 64 * NPTS;

  _Float16 ch[96];
  ch[0] = (_Float16)(Xz[id] - newxyz[(b * 3 + 0) * NPOINT + s]);
  ch[1] = (_Float16)(Xz[NPTS + id] - newxyz[(b * 3 + 1) * NPOINT + s]);
  ch[2] = (_Float16)(Xz[2 * NPTS + id] - newxyz[(b * 3 + 2) * NPOINT + s]);
#pragma unroll
  for (int c = 0; c < 64; c++) ch[3 + c] = (_Float16)P[(size_t)c * NPTS + id];
#pragma unroll
  for (int c = 67; c < 96; c++) ch[c] = (_Float16)0.0f;

  _Float16* dst = feat + (size_t)gid * 96;
#pragma unroll
  for (int v = 0; v < 12; v++) {
    v8h t;
#pragma unroll
    for (int j = 0; j < 8; j++) t[j] = ch[v * 8 + j];
    *(v8h*)(dst + v * 8) = t;
  }
}

// ---------------------------------------------------------------------------
// GEMM: one wave owns a 16-row weight stripe (A fragments register-resident)
// and sweeps NTILE 16-column tiles. Optional fused BN+relu (packed f16) on
// the B operand of the *previous* layer's output. All loads b128-aligned.
// ---------------------------------------------------------------------------
template <int KPAD, bool HASBN>
__global__ void __launch_bounds__(256)
wmma_gemm(const _Float16* __restrict__ Xf, const _Float16* __restrict__ Wh,
          const float* __restrict__ bias, const _Float16* __restrict__ bnsc,
          const _Float16* __restrict__ bnsh, _Float16* __restrict__ Y,
          int Cout, int mtiles) {
  constexpr int NCHUNK = KPAD / 32;
  constexpr int NTILE = 4;               // 64 columns per wave
  const int TGPB = (TOTCOL / 16) / NTILE;  // 256 tile-groups per batch
  const int lane = threadIdx.x & 31;
  const int wid  = blockIdx.x * 8 + (threadIdx.x >> 5);
  const int mt   = wid % mtiles;
  const int rest = wid / mtiles;
  const int tg = rest % TGPB;
  const int b  = rest / TGPB;
  const int half = lane >> 4;
  const int l    = lane & 15;
  const int m    = mt * 16 + l;

  // A (weight) fragments: loaded once, reused for all NTILE tiles.
  v16h afrag[NCHUNK];
  const _Float16* wbase = Wh + (size_t)m * KPAD + half * 16;
#pragma unroll
  for (int ch = 0; ch < NCHUNK; ch++) {
    v8h a0 = *(const v8h*)(wbase + ch * 32);       // K = cc + half*16 .. +7
    v8h a1 = *(const v8h*)(wbase + ch * 32 + 8);   // K = cc+half*16+8 .. +15
    afrag[ch] = __builtin_shufflevector(a0, a1, 0, 1, 2, 3, 4, 5, 6, 7, 8, 9,
                                        10, 11, 12, 13, 14, 15);
  }

  // BN scale/shift fragments (per-K, packed f16): hoisted out of tile loop.
  v8h scf[NCHUNK][2], shf[NCHUNK][2];
  if (HASBN) {
#pragma unroll
    for (int ch = 0; ch < NCHUNK; ch++) {
      const _Float16* sp = bnsc + ch * 32 + half * 16;
      const _Float16* tp = bnsh + ch * 32 + half * 16;
      scf[ch][0] = *(const v8h*)sp;       scf[ch][1] = *(const v8h*)(sp + 8);
      shf[ch][0] = *(const v8h*)tp;       shf[ch][1] = *(const v8h*)(tp + 8);
    }
  }

  const int col0 = tg * (NTILE * 16) + l;
  const _Float16* xb = Xf + ((size_t)b * TOTCOL + col0) * KPAD + half * 16;
  const int obase = mt * 16 + half * 8;
  const float4* bp = (const float4*)(bias + obase);
  const float4 bv0 = bp[0], bv1 = bp[1];
  _Float16* yb = Y + ((size_t)b * TOTCOL + col0) * Cout + obase;

#pragma unroll
  for (int t = 0; t < NTILE; t++) {
    const _Float16* xt = xb + (size_t)t * 16 * KPAD;
    v8f acc = {};
#pragma unroll
    for (int ch = 0; ch < NCHUNK; ch++) {
      v8h b0 = *(const v8h*)(xt + ch * 32);      // K = cc + half*16 .. +15
      v8h b1 = *(const v8h*)(xt + ch * 32 + 8);
      if (HASBN) {
        b0 = b0 * scf[ch][0] + shf[ch][0];       // packed f16 fma
        b1 = b1 * scf[ch][1] + shf[ch][1];
#pragma unroll
        for (int j = 0; j < 8; j++) {            // relu (packed f16 max)
          b0[j] = (b0[j] > (_Float16)0.0f) ? b0[j] : (_Float16)0.0f;
          b1[j] = (b1[j] > (_Float16)0.0f) ? b1[j] : (_Float16)0.0f;
        }
      }
      v16h bm = __builtin_shufflevector(b0, b1, 0, 1, 2, 3, 4, 5, 6, 7, 8, 9,
                                        10, 11, 12, 13, 14, 15);
      acc = __builtin_amdgcn_wmma_f32_16x16x32_f16(false, afrag[ch], false, bm,
                                                   (short)0, acc, false, false);
    }
    v8h d;
    d[0] = (_Float16)(acc[0] + bv0.x);
    d[1] = (_Float16)(acc[1] + bv0.y);
    d[2] = (_Float16)(acc[2] + bv0.z);
    d[3] = (_Float16)(acc[3] + bv0.w);
    d[4] = (_Float16)(acc[4] + bv1.x);
    d[5] = (_Float16)(acc[5] + bv1.y);
    d[6] = (_Float16)(acc[6] + bv1.z);
    d[7] = (_Float16)(acc[7] + bv1.w);
    *(v8h*)(yb + (size_t)t * 16 * Cout) = d;     // one b128 store per tile
  }
}

// ---------------------------------------------------------------------------
// BN stats over f16 Y in (col, C) layout: coalesced v8h loads, register
// partials, one LDS combine, C global atomics per block.
// ---------------------------------------------------------------------------
template <int C>
__global__ void __launch_bounds__(256)
stats_kernel(const _Float16* __restrict__ Y, float* __restrict__ sum,
             float* __restrict__ sumsq) {
  const int G = C / 8;
  const int t = threadIdx.x;
  const int g = t % G;
  const int sub = t / G;
  const int colsPar = 256 / G;
  const int COLS_PER_BLOCK = 1024;   // total cols = 131072 -> 128 blocks
  const int col0 = blockIdx.x * COLS_PER_BLOCK;

  float s[8], q[8];
#pragma unroll
  for (int j = 0; j < 8; j++) { s[j] = 0.0f; q[j] = 0.0f; }

  for (int c = col0 + sub; c < col0 + COLS_PER_BLOCK; c += colsPar) {
    v8h v = *(const v8h*)(Y + (size_t)c * C + g * 8);
#pragma unroll
    for (int j = 0; j < 8; j++) {
      float f = (float)v[j];
      s[j] += f; q[j] += f * f;
    }
  }
  __shared__ float ls[C], lq[C];
  if (t < C) { ls[t] = 0.0f; lq[t] = 0.0f; }
  __syncthreads();
#pragma unroll
  for (int j = 0; j < 8; j++) {
    atomicAdd(&ls[g * 8 + j], s[j]);
    atomicAdd(&lq[g * 8 + j], q[j]);
  }
  __syncthreads();
  if (t < C) { atomicAdd(&sum[t], ls[t]); atomicAdd(&sumsq[t], lq[t]); }
}

__global__ void bn_finalize(const float* __restrict__ sum,
                            const float* __restrict__ sumsq,
                            const float* __restrict__ g,
                            const float* __restrict__ beta,
                            float* __restrict__ sc, float* __restrict__ sh,
                            _Float16* __restrict__ scH,
                            _Float16* __restrict__ shH, int C) {
  int c = threadIdx.x;
  if (c < C) {
    const float cnt = (float)BATCH * (float)TOTCOL;
    float mean = sum[c] / cnt;
    float var  = sumsq[c] / cnt - mean * mean;
    float s = g[c] * rsqrtf(var + BNEPS);
    float t = beta[c] - mean * s;
    sc[c] = s;  sh[c] = t;
    scH[c] = (_Float16)s;  shH[c] = (_Float16)t;
  }
}

__global__ void zero_stats(float* __restrict__ p) {
  int i = blockIdx.x * 256 + threadIdx.x;
  if (i < 768) p[i] = 0.0f;
}

// ---------------------------------------------------------------------------
// Final BN+relu+max over 32 neighbors; y2 in (col,128) f16, out (B,128,S) f32.
// ---------------------------------------------------------------------------
__global__ void __launch_bounds__(256)
maxpool_kernel(const _Float16* __restrict__ Y2, const float* __restrict__ sc,
               const float* __restrict__ sh, float* __restrict__ out) {
  int gid = blockIdx.x * 256 + threadIdx.x;   // b*65536 + s*128 + o
  int o = gid & 127;
  int s = (gid >> 7) & (NPOINT - 1);
  int b = gid >> 16;
  const _Float16* p = Y2 + ((size_t)b * TOTCOL + s * NSAMP) * 128 + o;
  float scale = sc[o], shift = sh[o];
  float m = 0.0f;
#pragma unroll
  for (int k = 0; k < NSAMP; k++)
    m = fmaxf(m, fmaxf((float)p[(size_t)k * 128] * scale + shift, 0.0f));
  out[((size_t)b * 128 + o) * NPOINT + s] = m;
}

// ---------------------------------------------------------------------------
extern "C" void kernel_launch(void* const* d_in, const int* in_sizes, int n_in,
                              void* d_out, int out_size, void* d_ws,
                              size_t ws_size, hipStream_t stream) {
  (void)in_sizes; (void)n_in; (void)out_size; (void)ws_size;
  const float* xyz = (const float*)d_in[0];
  const float* pts = (const float*)d_in[1];
  const float* W0  = (const float*)d_in[2];
  const float* b0  = (const float*)d_in[3];
  const float* g0  = (const float*)d_in[4];
  const float* be0 = (const float*)d_in[5];
  const float* W1  = (const float*)d_in[6];
  const float* b1  = (const float*)d_in[7];
  const float* g1  = (const float*)d_in[8];
  const float* be1 = (const float*)d_in[9];
  const float* W2  = (const float*)d_in[10];
  const float* b2  = (const float*)d_in[11];
  const float* g2  = (const float*)d_in[12];
  const float* be2 = (const float*)d_in[13];

  float* out = (float*)d_out;
  float* newxyz = out;                       // (B,3,S)
  float* pooled = out + BATCH * 3 * NPOINT;  // (B,128,S)

  char* ws = (char*)d_ws;
  size_t off = 0;
  int* gidx = (int*)(ws + off);              off += (size_t)BATCH * NPOINT * NSAMP * 4;
  int* fps_idx = (int*)(ws + off);           off += (size_t)BATCH * NPOINT * 4;
  float* stats = (float*)(ws + off);         off += 1536 * 4;
  float* sum0 = stats;        float* sq0 = stats + 128;
  float* sum1 = stats + 256;  float* sq1 = stats + 384;
  float* sum2 = stats + 512;  float* sq2 = stats + 640;
  float* sc0  = stats + 768;  float* sh0 = stats + 896;
  float* sc1  = stats + 1024; float* sh1 = stats + 1152;
  float* sc2  = stats + 1280; float* sh2 = stats + 1408;
  _Float16* bnH = (_Float16*)(ws + off);     off += 6 * 128 * 2;
  _Float16* sc0H = bnH;        _Float16* sh0H = bnH + 128;
  _Float16* sc1H = bnH + 256;  _Float16* sh1H = bnH + 384;
  _Float16* sc2H = bnH + 512;  _Float16* sh2H = bnH + 640;
  _Float16* Wh0 = (_Float16*)(ws + off);     off += 64 * 96 * 2;
  _Float16* Wh1 = (_Float16*)(ws + off);     off += 64 * 64 * 2;
  _Float16* Wh2 = (_Float16*)(ws + off);     off += 128 * 64 * 2;
  off = (off + 255) & ~(size_t)255;
  _Float16* feat = (_Float16*)(ws + off);    off += (size_t)BATCH * TOTCOL * 96 * 2;
  _Float16* y0 = (_Float16*)(ws + off);      off += (size_t)BATCH * TOTCOL * 64 * 2;
  _Float16* y1 = (_Float16*)(ws + off);      off += (size_t)BATCH * TOTCOL * 64 * 2;
  _Float16* y2 = (_Float16*)(ws + off);      off += (size_t)BATCH * TOTCOL * 128 * 2;

  zero_stats<<<3, 256, 0, stream>>>(stats);
  packW_kernel<<<(64 * 96 + 255) / 256, 256, 0, stream>>>(W0, Wh0, 64, 67, 96);
  packW_kernel<<<(64 * 64 + 255) / 256, 256, 0, stream>>>(W1, Wh1, 64, 64, 64);
  packW_kernel<<<(128 * 64 + 255) / 256, 256, 0, stream>>>(W2, Wh2, 128, 64, 64);

  fps_kernel<<<BATCH, 1024, 0, stream>>>(xyz, fps_idx, newxyz);
  ballquery_kernel<<<(BATCH * NPOINT) / 8, 256, 0, stream>>>(xyz, newxyz, gidx);
  gather_feat<<<(BATCH * TOTCOL) / 256, 256, 0, stream>>>(xyz, pts, gidx,
                                                          newxyz, feat);

  // layer 0: K=96 (padded 67) -> Cout=64, no BN on input
  wmma_gemm<96, false><<<1024, 256, 0, stream>>>(feat, Wh0, b0, sc0H, sh0H,
                                                 y0, 64, 4);
  stats_kernel<64><<<128, 256, 0, stream>>>(y0, sum0, sq0);
  bn_finalize<<<1, 128, 0, stream>>>(sum0, sq0, g0, be0, sc0, sh0, sc0H, sh0H,
                                     64);

  // layer 1: K=64 -> Cout=64, BN0+relu fused into B-operand
  wmma_gemm<64, true><<<1024, 256, 0, stream>>>(y0, Wh1, b1, sc0H, sh0H, y1,
                                                64, 4);
  stats_kernel<64><<<128, 256, 0, stream>>>(y1, sum1, sq1);
  bn_finalize<<<1, 128, 0, stream>>>(sum1, sq1, g1, be1, sc1, sh1, sc1H, sh1H,
                                     64);

  // layer 2: K=64 -> Cout=128, BN1+relu fused into B-operand
  wmma_gemm<64, true><<<2048, 256, 0, stream>>>(y1, Wh2, b2, sc1H, sh1H, y2,
                                                128, 8);
  stats_kernel<128><<<128, 256, 0, stream>>>(y2, sum2, sq2);
  bn_finalize<<<1, 128, 0, stream>>>(sum2, sq2, g2, be2, sc2, sh2, sc2H, sh2H,
                                     128);

  // BN2 + relu + max over K (fused into the pooling read)
  maxpool_kernel<<<(BATCH * 128 * NPOINT) / 256, 256, 0, stream>>>(y2, sc2,
                                                                   sh2, pooled);
}
